// HeteroGAT_44375602102316
// MI455X (gfx1250) — matrix operand order repaired and checked by reading, not compiled, not run on hardware
//
#include <hip/hip_runtime.h>

// ---------------------------------------------------------------------------
// HeteroGAT for MI455X (gfx1250, wave32). Dense projections via
// v_wmma_f32_16x16x32_bf16 (K=32 == HID, one WMMA per 16x16 tile, tile loop
// fully unrolled). W is pre-packed to bf16 in B-operand lane layout so the
// hot kernel does one 32B load per lane per tile. Gather/softmax/scatter via
// L2 float atomics (192MB L2 >> per-conv hot set).
// ---------------------------------------------------------------------------

#define NU 500000
#define NB 300000
#define NM 200000
#define EBN 1500000
#define EMN 1500000
#define HID 32
#define NEG_SLOPE 0.2f

typedef __attribute__((ext_vector_type(16))) __bf16 v16bf;
typedef __attribute__((ext_vector_type(8)))  float  v8f;

// ---------------- pack W[32,ncols] (f32, row-major) -> Bp[ntiles][32 lanes][16] bf16
// Lane l of tile t needs B[k, t*16 + (l&15)] for k = (l>=16 ? 8..15,24..31 : 0..7,16..23).
__global__ void pack_b_kernel(const float* __restrict__ W, __bf16* __restrict__ Bp, int ncols)
{
    int idx = blockIdx.x * blockDim.x + threadIdx.x;
    int total = (ncols >> 4) << 9;            // ntiles * 32 * 16
    if (idx >= total) return;
    int k16  = idx & 15;
    int lane = (idx >> 4) & 31;
    int t    = idx >> 9;
    int col  = (t << 4) + (lane & 15);
    int k    = ((lane >> 4) ? 8 : 0) + (k16 & 7) + ((k16 >> 3) << 4);
    Bp[idx] = (__bf16)W[(size_t)k * ncols + col];
}

// ---------------- WMMA projection: out[nrows,NCOLS] = relu?(x[nrows,32] @ W + b)
// One wave computes a 16-row strip; NCOLS/16 tiles fully unrolled.
template <int NCOLS>
__global__ void proj_wmma_kernel(const float* __restrict__ x,
                                 const __bf16* __restrict__ Bp,
                                 const float* __restrict__ bias,
                                 float* __restrict__ out,
                                 int nrows, int do_relu)
{
    const int lane = threadIdx.x & 31;
    const int wave = threadIdx.x >> 5;
    const int rowBase = (blockIdx.x * (blockDim.x >> 5) + wave) << 4;
    if (rowBase >= nrows) return;   // whole-wave uniform -> EXEC stays all-1s

    const int m  = lane & 15;
    const int hi = lane >> 4;       // 0: lanes 0-15, 1: lanes 16-31
    const int row = rowBase + m;

    // A operand 16x32 bf16 (ISA 7.12.2): lanes<16 hold K=0..7,16..23; lanes>=16 K=8..15,24..31
    v16bf a;
    {
        const int k0 = hi ? 8 : 0;
        const float4* r0 = (const float4*)(x + (size_t)row * HID + k0);       // K-run 0 (8 vals)
        const float4* r1 = (const float4*)(x + (size_t)row * HID + k0 + 16);  // K-run 1 (8 vals)
        float4 p0 = r0[0], p1 = r0[1], q0 = r1[0], q1 = r1[1];
        a[0]=(__bf16)p0.x; a[1]=(__bf16)p0.y; a[2]=(__bf16)p0.z; a[3]=(__bf16)p0.w;
        a[4]=(__bf16)p1.x; a[5]=(__bf16)p1.y; a[6]=(__bf16)p1.z; a[7]=(__bf16)p1.w;
        a[8]=(__bf16)q0.x; a[9]=(__bf16)q0.y; a[10]=(__bf16)q0.z; a[11]=(__bf16)q0.w;
        a[12]=(__bf16)q1.x; a[13]=(__bf16)q1.y; a[14]=(__bf16)q1.z; a[15]=(__bf16)q1.w;
    }

    constexpr int NT = NCOLS >> 4;
#pragma unroll
    for (int t = 0; t < NT; ++t) {
        const int col = (t << 4) + m;
        // one coalesced 32B load per lane: pre-packed B operand
        v16bf b = *(const v16bf*)(Bp + (((size_t)t << 5) + lane) * 16);

        v8f c = {};
        c = __builtin_amdgcn_wmma_f32_16x16x32_bf16(false, a, false, b, (short)0, c, false, false);

        const float bv = bias ? bias[col] : 0.0f;
#pragma unroll
        for (int r = 0; r < 8; ++r) {       // D layout: VGPR r -> M = r + 8*hi, N = lane&15
            float v = c[r] + bv;
            if (do_relu) v = v > 0.0f ? v : 0.0f;
            out[(size_t)(rowBase + r + (hi ? 8 : 0)) * NCOLS + col] = v;
        }
    }
}

// ---------------- watt[k,h] = sum_c W[k, h*32+c] * att[h*32+c]   (W is 32x64)
__global__ void make_watt_kernel(const float* __restrict__ W,
                                 const float* __restrict__ att,
                                 float* __restrict__ watt)
{
    int t = threadIdx.x;
    if (t >= 64) return;
    int k = t >> 1, h = t & 1;
    float s = 0.f;
#pragma unroll
    for (int c = 0; c < HID; ++c) s += W[k * 64 + h * HID + c] * att[h * HID + c];
    watt[k * 2 + h] = s;
}

// ---------------- a[i,h] = x[i,:] @ watt[:,h]
__global__ void att_proj_kernel(const float* __restrict__ x,
                                const float* __restrict__ watt,
                                float* __restrict__ a, int n)
{
    int i = blockIdx.x * blockDim.x + threadIdx.x;
    if (i >= n) return;
    const float* xr = x + (size_t)i * HID;
    float s0 = 0.f, s1 = 0.f;
#pragma unroll
    for (int k = 0; k < HID; ++k) {
        float v = xr[k];
        s0 += v * watt[k * 2 + 0];
        s1 += v * watt[k * 2 + 1];
    }
    a[(size_t)i * 2 + 0] = s0;
    a[(size_t)i * 2 + 1] = s1;
}

__global__ void fill_u32_kernel(unsigned* __restrict__ p, unsigned v, size_t n)
{
    size_t i = (size_t)blockIdx.x * blockDim.x + threadIdx.x;
    if (i < n) p[i] = v;
}

// sortable-uint mapping so unsigned atomicMax == float max
__device__ __forceinline__ float    lrelu(float a) { return a > 0.f ? a : NEG_SLOPE * a; }
__device__ __forceinline__ unsigned f2s(float f)   { unsigned u = __float_as_uint(f); return (u & 0x80000000u) ? ~u : (u | 0x80000000u); }
__device__ __forceinline__ float    s2f(unsigned s){ unsigned u = (s & 0x80000000u) ? (s & 0x7FFFFFFFu) : ~s; return __uint_as_float(u); }

__global__ void edge_max_kernel(const int* __restrict__ src, const int* __restrict__ dst,
                                const float* __restrict__ a_src, const float* __restrict__ a_dst,
                                unsigned* __restrict__ m, int E)
{
    int e = blockIdx.x * blockDim.x + threadIdx.x;
    if (e >= E) return;
    int s = src[e], d = dst[e];
#pragma unroll
    for (int h = 0; h < 2; ++h) {
        float a = lrelu(a_src[(size_t)s * 2 + h] + a_dst[(size_t)d * 2 + h]);
        atomicMax(&m[(size_t)d * 2 + h], f2s(a));
    }
}

// in-place: sortable-u32 -> float, empty segments (-inf) / nonfinite -> 0
__global__ void fin_m_kernel(unsigned* __restrict__ m, int n)
{
    int i = blockIdx.x * blockDim.x + threadIdx.x;
    if (i >= n) return;
    float f = s2f(m[i]);
    unsigned u = __float_as_uint(f);
    if ((u & 0x7F800000u) == 0x7F800000u) f = 0.f;
    ((float*)m)[i] = f;
}

__global__ void edge_exp_kernel(const int* __restrict__ src, const int* __restrict__ dst,
                                const float* __restrict__ a_src, const float* __restrict__ a_dst,
                                const float* __restrict__ m, float* __restrict__ ealpha,
                                float* __restrict__ denom, int E)
{
    int e = blockIdx.x * blockDim.x + threadIdx.x;
    if (e >= E) return;
    int s = src[e], d = dst[e];
#pragma unroll
    for (int h = 0; h < 2; ++h) {
        float a  = lrelu(a_src[(size_t)s * 2 + h] + a_dst[(size_t)d * 2 + h]);
        float ee = __expf(a - m[(size_t)d * 2 + h]);
        ealpha[(size_t)e * 2 + h] = ee;
        atomicAdd(&denom[(size_t)d * 2 + h], ee);
    }
}

// one wave per edge: lane l handles channel l (head 0) and 32+l (head 1)
__global__ void edge_scatter_kernel(const int* __restrict__ src, const int* __restrict__ dst,
                                    const float* __restrict__ ealpha, const float* __restrict__ denom,
                                    const float* __restrict__ hs, float* __restrict__ acc, int E)
{
    long long tid = (long long)blockIdx.x * blockDim.x + threadIdx.x;
    int e    = (int)(tid >> 5);
    int lane = (int)(tid & 31);
    if (e >= E) return;
    int s = src[e], d = dst[e];
    float al0 = ealpha[(size_t)e * 2 + 0] / (denom[(size_t)d * 2 + 0] + 1e-16f);
    float al1 = ealpha[(size_t)e * 2 + 1] / (denom[(size_t)d * 2 + 1] + 1e-16f);
    const float* hsr = hs  + (size_t)s * 64;
    float*       ar  = acc + (size_t)d * 64;
    atomicAdd(&ar[lane],      al0 * hsr[lane]);
    atomicAdd(&ar[32 + lane], al1 * hsr[32 + lane]);
}

// head-mean + bias(es) (+ optional relu) -> out[n,32]
__global__ void finalize_kernel(const float* __restrict__ acc,
                                const float* __restrict__ b1, const float* __restrict__ b2,
                                float* __restrict__ out, int n, int do_relu)
{
    int t = blockIdx.x * blockDim.x + threadIdx.x;
    if (t >= n * HID) return;
    int i = t >> 5, c = t & 31;
    float v = 0.5f * (acc[(size_t)i * 64 + c] + acc[(size_t)i * 64 + 32 + c]) + b1[c];
    if (b2) v += b2[c];
    if (do_relu) v = v > 0.f ? v : 0.f;
    out[(size_t)i * HID + c] = v;
}

// ---------------------------------------------------------------------------
extern "C" void kernel_launch(void* const* d_in, const int* in_sizes, int n_in,
                              void* d_out, int out_size, void* d_ws, size_t ws_size,
                              hipStream_t stream)
{
    (void)in_sizes; (void)n_in; (void)out_size; (void)ws_size;

    // --- input map (setup_inputs() dict insertion order, flattened) ---
    const float* emb_u = (const float*)d_in[0];
    const float* emb_b = (const float*)d_in[1];
    const float* emb_m = (const float*)d_in[2];
    // conv1: idx 3..22   conv2: idx 23..42   (rel order u2b,b2u,u2m,m2u; item Ws,Wd,as,ad,b)
    // lin: 43..48        edges: 49 (rb), 50 (rm)
    const int* erb = (const int*)d_in[49];
    const int* erm = (const int*)d_in[50];
    const int* rb_u = erb;             // row 0: user ids
    const int* rb_b = erb + EBN;       // row 1: book ids
    const int* rm_u = erm;
    const int* rm_m = erm + EMN;

    auto P = [&](int cb, int rel, int item) -> const float* {
        return (const float*)d_in[cb + rel * 5 + item];
    };

    // --- workspace carve-out ---
    char* base = (char*)d_ws;
    size_t off = 0;
    auto alloc = [&](size_t nfl) -> float* {
        float* p = (float*)(base + off);
        off += ((nfl * sizeof(float) + 255) / 256) * 256;
        return p;
    };
    float* feat_u = alloc((size_t)NU * HID);
    float* feat_b = alloc((size_t)NB * HID);
    float* feat_m = alloc((size_t)NM * HID);
    float* nfu    = alloc((size_t)NU * HID);
    float* nfb    = alloc((size_t)NB * HID);
    float* nfm    = alloc((size_t)NM * HID);
    float* hs     = alloc((size_t)NU * 64);    // src projection (max Ns = NU)
    float* acc    = alloc((size_t)NU * 64);    // per-head accumulation (max Nd = NU)
    float* a_src  = alloc((size_t)NU * 2);
    float* a_dst  = alloc((size_t)NU * 2);
    float* mbuf   = alloc((size_t)NU * 2);     // sortable-u32 then float, in place
    float* denom  = alloc((size_t)NU * 2);
    float* ealpha = alloc((size_t)EBN * 2);
    float* watt_s = alloc(64);
    float* watt_d = alloc(64);
    __bf16* bpack = (__bf16*)alloc(2048);      // packed B operand (max 32x64 bf16 = 4KB)

    auto cdiv = [](long long a, long long b) { return (int)((a + b - 1) / b); };

    auto proj64 = [&](const float* x, const float* W, float* out, int nrows) {
        pack_b_kernel<<<8, 256, 0, stream>>>(W, bpack, 64);
        proj_wmma_kernel<64><<<cdiv(nrows, 128), 256, 0, stream>>>(x, bpack, nullptr, out, nrows, 0);
    };
    auto proj32 = [&](const float* x, const float* W, const float* bias, float* out, int nrows) {
        pack_b_kernel<<<4, 256, 0, stream>>>(W, bpack, 32);
        proj_wmma_kernel<32><<<cdiv(nrows, 128), 256, 0, stream>>>(x, bpack, bias, out, nrows, 1);
    };
    auto fill = [&](void* p, unsigned v, size_t n) {
        fill_u32_kernel<<<cdiv((long long)n, 256), 256, 0, stream>>>((unsigned*)p, v, n);
    };

    const unsigned SORT_NEG_INF = 0x007FFFFFu;   // f2s(-inf)

    auto run_gat = [&](const float* xs, int Ns, const float* xd, int Nd,
                       const int* src, const int* dst, int E,
                       const float* Ws, const float* Wd, const float* as_, const float* ad_,
                       bool zero_acc) {
        make_watt_kernel<<<1, 64, 0, stream>>>(Ws, as_, watt_s);
        make_watt_kernel<<<1, 64, 0, stream>>>(Wd, ad_, watt_d);
        proj64(xs, Ws, hs, Ns);                                                     // hs = xs @ Ws
        att_proj_kernel<<<cdiv(Ns, 256), 256, 0, stream>>>(xs, watt_s, a_src, Ns);  // a_src
        att_proj_kernel<<<cdiv(Nd, 256), 256, 0, stream>>>(xd, watt_d, a_dst, Nd);  // a_dst
        fill(mbuf, SORT_NEG_INF, (size_t)Nd * 2);
        fill(denom, 0u, (size_t)Nd * 2);
        if (zero_acc) fill(acc, 0u, (size_t)Nd * 64);
        edge_max_kernel<<<cdiv(E, 256), 256, 0, stream>>>(src, dst, a_src, a_dst, (unsigned*)mbuf, E);
        fin_m_kernel<<<cdiv((long long)Nd * 2, 256), 256, 0, stream>>>((unsigned*)mbuf, Nd * 2);
        edge_exp_kernel<<<cdiv(E, 256), 256, 0, stream>>>(src, dst, a_src, a_dst,
                                                          (const float*)mbuf, ealpha, denom, E);
        edge_scatter_kernel<<<cdiv((long long)E * 32, 256), 256, 0, stream>>>(src, dst, ealpha,
                                                                              denom, hs, acc, E);
    };

    auto layer = [&](int cb, const float* xu, const float* xb, const float* xm,
                     float* ou, float* ob, float* om, int relu) {
        // book <- u2b
        run_gat(xu, NU, xb, NB, rb_u, rb_b, EBN, P(cb,0,0), P(cb,0,1), P(cb,0,2), P(cb,0,3), true);
        finalize_kernel<<<cdiv((long long)NB * HID, 256), 256, 0, stream>>>(acc, P(cb,0,4), nullptr, ob, NB, relu);
        // movie <- u2m
        run_gat(xu, NU, xm, NM, rm_u, rm_m, EMN, P(cb,2,0), P(cb,2,1), P(cb,2,2), P(cb,2,3), true);
        finalize_kernel<<<cdiv((long long)NM * HID, 256), 256, 0, stream>>>(acc, P(cb,2,4), nullptr, om, NM, relu);
        // user <- b2u + m2u (linear head-mean -> share one accumulator)
        run_gat(xb, NB, xu, NU, rb_b, rb_u, EBN, P(cb,1,0), P(cb,1,1), P(cb,1,2), P(cb,1,3), true);
        run_gat(xm, NM, xu, NU, rm_m, rm_u, EMN, P(cb,3,0), P(cb,3,1), P(cb,3,2), P(cb,3,3), false);
        finalize_kernel<<<cdiv((long long)NU * HID, 256), 256, 0, stream>>>(acc, P(cb,1,4), P(cb,3,4), ou, NU, relu);
    };

    layer(3,  emb_u,  emb_b,  emb_m,  feat_u, feat_b, feat_m, 1);  // conv1 + relu
    layer(23, feat_u, feat_b, feat_m, nfu,    nfb,    nfm,    0);  // conv2

    // final per-type linear + relu, straight into d_out (u | b | m)
    float* out_u = (float*)d_out;
    float* out_b = out_u + (size_t)NU * HID;
    float* out_m = out_b + (size_t)NB * HID;
    proj32(nfu, (const float*)d_in[43], (const float*)d_in[44], out_u, NU);
    proj32(nfb, (const float*)d_in[45], (const float*)d_in[46], out_b, NB);
    proj32(nfm, (const float*)d_in[47], (const float*)d_in[48], out_m, NM);
}
